// FeatureEncoderCE_47536698032403
// MI455X (gfx1250) — matrix operand order repaired and checked by reading, not compiled, not run on hardware
//
#include <hip/hip_runtime.h>
#include <hip/hip_bf16.h>
#include <cstdint>
#include <cstddef>

// ---------------------------------------------------------------------------
// Problem constants (from reference): BS=256, T=128, D=HID=256, N=15 negatives
// ---------------------------------------------------------------------------
#define BS   256
#define TT   128
#define DD   256
#define HID  256
#define NNEG 15

typedef __attribute__((ext_vector_type(16))) __bf16 v16bf;
typedef __attribute__((ext_vector_type(8)))  float  v8f;

// ---------------------------------------------------------------------------
// Fragment loaders for CDNA5 WMMA (wave32), k order k = 16*khalf + s in both A and B.
// Base pointers are per-lane and already include the lane's k-half offset, so
// per-kstep addresses are base + ks*64B (folds to immediate offsets).
// A (16x32 bf16): lane m=lane&15, khalf=lane>>4; base = row + khalf*16 elems;
//                 q0 = +ks*32 elems, q1 = q0 + 8 elems.
// B (32x16 bf16): lane n=lane&15; base = Wrow(n) + (lane>>4)*16 elems;
//                 q0 = +ks*32 elems, q1 = q0 + 8 elems.
// ---------------------------------------------------------------------------
__device__ __forceinline__ v16bf load_frag_a_pre(const __hip_bfloat16* base, int ks) {
    union { uint4 q[2]; v16bf v; } u;
    const char* p = (const char*)(base + ks * 32);
    u.q[0] = *(const uint4*)(p);
    u.q[1] = *(const uint4*)(p + 16);
    return u.v;
}

__device__ __forceinline__ v16bf load_frag_b_pre(const __hip_bfloat16* base, int ks) {
    union { uint4 q[2]; v16bf v; } u;
    const char* p = (const char*)(base + ks * 32);
    u.q[0] = *(const uint4*)(p);
    u.q[1] = *(const uint4*)(p + 16);
    return u.v;
}

__device__ __forceinline__ v8f wmma_bf16(v16bf a, v16bf b, v8f c) {
    return __builtin_amdgcn_wmma_f32_16x16x32_bf16(false, a, false, b, (short)0, c, false, false);
}

__device__ __forceinline__ float sigmoid_fast(float x) { return 1.f / (1.f + __expf(-x)); }
__device__ __forceinline__ float tanh_fast(float x)    { return 1.f - 2.f / (__expf(2.f * x) + 1.f); }
__device__ __forceinline__ float gelu_exact(float x)   { return 0.5f * x * (1.f + erff(x * 0.70710678118654752f)); }

// LDS byte-offset of a generic pointer known to point into shared memory.
__device__ __forceinline__ unsigned lds_offset_of(const void* p) {
    return (unsigned)(uintptr_t)(__attribute__((address_space(3))) const void*)p;
}

// Per-lane async copy of one 16-byte chunk: global -> LDS (ASYNCcnt-tracked).
__device__ __forceinline__ void async_b128_to_lds(unsigned lds_off, const void* gaddr) {
    asm volatile("global_load_async_to_lds_b128 %0, %1, off" :: "v"(lds_off), "v"(gaddr) : "memory");
}
__device__ __forceinline__ void wait_asynccnt0() {
    asm volatile("s_wait_asynccnt 0x0" ::: "memory");
}

// ---------------------------------------------------------------------------
// fp32 -> bf16 conversion (weights + P), grid-stride
// ---------------------------------------------------------------------------
__global__ void cvt_f32_to_bf16(const float* __restrict__ in, __hip_bfloat16* __restrict__ out, int n) {
    for (int i = blockIdx.x * blockDim.x + threadIdx.x; i < n; i += gridDim.x * blockDim.x)
        out[i] = __float2bfloat16(in[i]);
}

__global__ void zero_scalar(float* p) { if (threadIdx.x == 0 && blockIdx.x == 0) *p = 0.f; }

// ---------------------------------------------------------------------------
// Persistent GRU layer. One WG per 16-row batch tile; 8 waves; wave w owns
// h columns [w*32, w*32+32). h lives in registers across all T steps; bf16 h_t
// is staged in LDS for next step's h@Whh^T A-fragments. The x_t tile (8 KB)
// is double-buffered in LDS via global_load_async_to_lds_b128 (prefetch t+1
// while computing t), removing the 8x redundant per-wave x loads.
// ---------------------------------------------------------------------------
__global__ __launch_bounds__(256) void gru_layer_kernel(
    const __hip_bfloat16* __restrict__ Xbf,    // (BS,T,256) bf16
    const __hip_bfloat16* __restrict__ Wihbf,  // (768,256) bf16 row-major
    const __hip_bfloat16* __restrict__ Whhbf,  // (768,256) bf16 row-major
    const float* __restrict__ bih,
    const float* __restrict__ bhh,
    __hip_bfloat16* __restrict__ Houtbf,       // (BS,T,HID) bf16
    float* __restrict__ Houtf)                 // optional (BS,T,HID) fp32
{
    __shared__ alignas(16) __hip_bfloat16 hbuf[16 * HID];
    __shared__ alignas(16) __hip_bfloat16 xbuf[2][16 * DD];

    const int tid   = threadIdx.x;
    const int lane  = tid & 31;
    const int btile = blockIdx.x;              // 16 tiles of 16 batch rows
    const int m     = lane & 15;
    const int khalf = lane >> 4;
    const int jbase = (tid >> 5) * 32;

    for (int i = tid; i < 16 * HID; i += 256) hbuf[i] = __float2bfloat16(0.f);

    // per-lane gate biases (n = lane&15 constant across a lane's C VGPRs)
    float b_r[2], b_z[2], b_in[2], b_hn[2];
    #pragma unroll
    for (int nt = 0; nt < 2; ++nt) {
        const int col = jbase + nt * 16 + m;
        b_r[nt]  = bih[col]           + bhh[col];
        b_z[nt]  = bih[HID + col]     + bhh[HID + col];
        b_in[nt] = bih[2 * HID + col];
        b_hn[nt] = bhh[2 * HID + col];
    }

    // per-lane weight base pointers (include k-half offset); invariant over t
    const __hip_bfloat16* wb[12];
    #pragma unroll
    for (int nt = 0; nt < 2; ++nt) {
        const int ncol = jbase + nt * 16 + m;
        wb[nt * 6 + 0] = Wihbf + (size_t)(ncol) * DD            + khalf * 16;
        wb[nt * 6 + 1] = Whhbf + (size_t)(ncol) * HID           + khalf * 16;
        wb[nt * 6 + 2] = Wihbf + (size_t)(HID + ncol) * DD      + khalf * 16;
        wb[nt * 6 + 3] = Whhbf + (size_t)(HID + ncol) * HID     + khalf * 16;
        wb[nt * 6 + 4] = Wihbf + (size_t)(2 * HID + ncol) * DD  + khalf * 16;
        wb[nt * 6 + 5] = Whhbf + (size_t)(2 * HID + ncol) * HID + khalf * 16;
    }

    // async-stage x_0 tile into xbuf[0]: 512 x b128 chunks, 2 per thread
    {
        #pragma unroll
        for (int r = 0; r < 2; ++r) {
            const int c    = tid + r * 256;
            const int row  = c >> 5;
            const int ch   = c & 31;
            const char* g  = (const char*)Xbf + ((size_t)(btile * 16 + row) * TT + 0) * (DD * 2) + ch * 16;
            const unsigned l = lds_offset_of((const char*)&xbuf[0][0] + row * (DD * 2) + ch * 16);
            async_b128_to_lds(l, g);
        }
    }

    v8f hreg[2] = {};

    for (int t = 0; t < TT; ++t) {
        wait_asynccnt0();      // this wave's async stores to xbuf[t&1] done
        __syncthreads();       // all waves' asyncs done + hbuf(h_{t-1}) visible

        // prefetch x_{t+1} into the other buffer (overlaps with compute)
        if (t + 1 < TT) {
            #pragma unroll
            for (int r = 0; r < 2; ++r) {
                const int c    = tid + r * 256;
                const int row  = c >> 5;
                const int ch   = c & 31;
                const char* g  = (const char*)Xbf + ((size_t)(btile * 16 + row) * TT + (t + 1)) * (DD * 2) + ch * 16;
                const unsigned l = lds_offset_of((const char*)&xbuf[(t + 1) & 1][0] + row * (DD * 2) + ch * 16);
                async_b128_to_lds(l, g);
            }
        }

        const __hip_bfloat16* xa_base = &xbuf[t & 1][m * DD] + khalf * 16;
        const __hip_bfloat16* ha_base = &hbuf[m * HID] + khalf * 16;

        v8f acc_r[2]  = {};
        v8f acc_z[2]  = {};
        v8f acc_in[2] = {};
        v8f acc_hn[2] = {};

        for (int ks = 0; ks < DD / 32; ++ks) {
            const v16bf a_x = load_frag_a_pre(xa_base, ks);
            const v16bf a_h = load_frag_a_pre(ha_base, ks);
            // 12 distinct temporaries: lets the scheduler clause the loads and
            // overlap them with the WMMAs (decremental s_wait_loadcnt).
            const v16bf B0  = load_frag_b_pre(wb[0],  ks);
            const v16bf B1  = load_frag_b_pre(wb[1],  ks);
            const v16bf B2  = load_frag_b_pre(wb[2],  ks);
            const v16bf B3  = load_frag_b_pre(wb[3],  ks);
            const v16bf B4  = load_frag_b_pre(wb[4],  ks);
            const v16bf B5  = load_frag_b_pre(wb[5],  ks);
            const v16bf B6  = load_frag_b_pre(wb[6],  ks);
            const v16bf B7  = load_frag_b_pre(wb[7],  ks);
            const v16bf B8  = load_frag_b_pre(wb[8],  ks);
            const v16bf B9  = load_frag_b_pre(wb[9],  ks);
            const v16bf B10 = load_frag_b_pre(wb[10], ks);
            const v16bf B11 = load_frag_b_pre(wb[11], ks);

            acc_r[0]  = wmma_bf16(a_x, B0,  acc_r[0]);
            acc_r[0]  = wmma_bf16(a_h, B1,  acc_r[0]);
            acc_z[0]  = wmma_bf16(a_x, B2,  acc_z[0]);
            acc_z[0]  = wmma_bf16(a_h, B3,  acc_z[0]);
            acc_in[0] = wmma_bf16(a_x, B4,  acc_in[0]);
            acc_hn[0] = wmma_bf16(a_h, B5,  acc_hn[0]);
            acc_r[1]  = wmma_bf16(a_x, B6,  acc_r[1]);
            acc_r[1]  = wmma_bf16(a_h, B7,  acc_r[1]);
            acc_z[1]  = wmma_bf16(a_x, B8,  acc_z[1]);
            acc_z[1]  = wmma_bf16(a_h, B9,  acc_z[1]);
            acc_in[1] = wmma_bf16(a_x, B10, acc_in[1]);
            acc_hn[1] = wmma_bf16(a_h, B11, acc_hn[1]);
        }
        __syncthreads();   // all waves done reading h_{t-1} fragments

        // gates + state update (lane-wise on matching C layouts)
        #pragma unroll
        for (int nt = 0; nt < 2; ++nt) {
            #pragma unroll
            for (int e = 0; e < 8; ++e) {
                const float r = sigmoid_fast(acc_r[nt][e] + b_r[nt]);
                const float z = sigmoid_fast(acc_z[nt][e] + b_z[nt]);
                const float n = tanh_fast(acc_in[nt][e] + b_in[nt] + r * (acc_hn[nt][e] + b_hn[nt]));
                hreg[nt][e] = (1.f - z) * n + z * hreg[nt][e];
            }
        }

        // publish h_t (bf16 to LDS for next step, bf16+fp32 to global)
        #pragma unroll
        for (int nt = 0; nt < 2; ++nt) {
            const int col = jbase + nt * 16 + m;
            #pragma unroll
            for (int e = 0; e < 8; ++e) {
                const int row = e + 8 * khalf;            // C layout: M = v + 8*(lane>=16)
                const __hip_bfloat16 hb = __float2bfloat16(hreg[nt][e]);
                hbuf[row * HID + col] = hb;
                const size_t g = ((size_t)(btile * 16 + row) * TT + t) * HID + col;
                Houtbf[g] = hb;
                if (Houtf) Houtf[g] = hreg[nt][e];
            }
        }
        // next iteration's (wait_asynccnt + __syncthreads) orders hbuf/xbuf use
    }
}

// ---------------------------------------------------------------------------
// out[row, n] = sum_k A[row,k] * W[n,k], rows flat over BS*T, 16 rows/WG.
// Used for WC = H @ W^T (fp32 result into workspace).
// ---------------------------------------------------------------------------
__global__ __launch_bounds__(256) void gemm_nt_kernel(
    const __hip_bfloat16* __restrict__ Abf,   // (rows, 256)
    const __hip_bfloat16* __restrict__ Wbf,   // (256, 256) row-major
    float* __restrict__ outF)                 // (rows, 256)
{
    const int lane  = threadIdx.x & 31;
    const int m     = lane & 15;
    const int khalf = lane >> 4;
    const int jbase = (threadIdx.x >> 5) * 32;
    const int rbase = blockIdx.x * 16;

    const __hip_bfloat16* arow = Abf + (size_t)(rbase + m) * DD + khalf * 16;
    const __hip_bfloat16* wb0  = Wbf + (size_t)(jbase + m) * DD + khalf * 16;
    const __hip_bfloat16* wb1  = wb0 + (size_t)16 * DD;

    v8f acc[2] = {};
    for (int ks = 0; ks < DD / 32; ++ks) {
        const v16bf a  = load_frag_a_pre(arow, ks);
        const v16bf b0 = load_frag_b_pre(wb0, ks);
        const v16bf b1 = load_frag_b_pre(wb1, ks);
        acc[0] = wmma_bf16(a, b0, acc[0]);
        acc[1] = wmma_bf16(a, b1, acc[1]);
    }
    #pragma unroll
    for (int nt = 0; nt < 2; ++nt) {
        const int col = jbase + nt * 16 + m;
        #pragma unroll
        for (int e = 0; e < 8; ++e) {
            const int row = e + 8 * khalf;
            outF[(size_t)(rbase + row) * DD + col] = acc[nt][e];
        }
    }
}

// ---------------------------------------------------------------------------
// R = gelu(H @ Wr1^T) @ Wr2^T  (fused two-stage, gelu result staged in LDS)
// ---------------------------------------------------------------------------
__global__ __launch_bounds__(256) void r_proj_kernel(
    const __hip_bfloat16* __restrict__ Hbf,
    const __hip_bfloat16* __restrict__ Wr1bf,
    const __hip_bfloat16* __restrict__ Wr2bf,
    float* __restrict__ Rout)
{
    __shared__ alignas(16) __hip_bfloat16 gbuf[16 * HID];

    const int lane  = threadIdx.x & 31;
    const int m     = lane & 15;
    const int khalf = lane >> 4;
    const int jbase = (threadIdx.x >> 5) * 32;
    const int rbase = blockIdx.x * 16;

    const __hip_bfloat16* hrow = Hbf + (size_t)(rbase + m) * HID + khalf * 16;

    // stage 1: G = gelu(H @ Wr1^T)
    {
        const __hip_bfloat16* wb0 = Wr1bf + (size_t)(jbase + m) * HID + khalf * 16;
        const __hip_bfloat16* wb1 = wb0 + (size_t)16 * HID;
        v8f acc[2] = {};
        for (int ks = 0; ks < HID / 32; ++ks) {
            const v16bf a  = load_frag_a_pre(hrow, ks);
            const v16bf b0 = load_frag_b_pre(wb0, ks);
            const v16bf b1 = load_frag_b_pre(wb1, ks);
            acc[0] = wmma_bf16(a, b0, acc[0]);
            acc[1] = wmma_bf16(a, b1, acc[1]);
        }
        #pragma unroll
        for (int nt = 0; nt < 2; ++nt) {
            const int col = jbase + nt * 16 + m;
            #pragma unroll
            for (int e = 0; e < 8; ++e) {
                const int row = e + 8 * khalf;
                gbuf[row * HID + col] = __float2bfloat16(gelu_exact(acc[nt][e]));
            }
        }
    }
    __syncthreads();

    // stage 2: R = G @ Wr2^T
    {
        const __hip_bfloat16* ga  = &gbuf[m * HID] + khalf * 16;
        const __hip_bfloat16* wb0 = Wr2bf + (size_t)(jbase + m) * HID + khalf * 16;
        const __hip_bfloat16* wb1 = wb0 + (size_t)16 * HID;
        v8f acc2[2] = {};
        for (int ks = 0; ks < HID / 32; ++ks) {
            const v16bf a  = load_frag_a_pre(ga, ks);
            const v16bf b0 = load_frag_b_pre(wb0, ks);
            const v16bf b1 = load_frag_b_pre(wb1, ks);
            acc2[0] = wmma_bf16(a, b0, acc2[0]);
            acc2[1] = wmma_bf16(a, b1, acc2[1]);
        }
        #pragma unroll
        for (int nt = 0; nt < 2; ++nt) {
            const int col = jbase + nt * 16 + m;
            #pragma unroll
            for (int e = 0; e < 8; ++e) {
                const int row = e + 8 * khalf;
                Rout[(size_t)(rbase + row) * DD + col] = acc2[nt][e];
            }
        }
    }
}

// ---------------------------------------------------------------------------
// Contrastive loss + alpha. One WG per (t,b); 16 block-reductions over D=256.
// logits[c] = mean_d WC[b,t,d] * P[row_c, t+1, d];  row_0=b, row_c=neg_idx.
// loss += -(logits[0]-lse)/(T-1); alpha[b] = logits[0] at t==T-2.
// ---------------------------------------------------------------------------
__global__ __launch_bounds__(256) void contrast_kernel(
    const float* __restrict__ WCf,       // (BS, T, D), only t<T-1 used
    const float* __restrict__ P,         // (BS, T, D)
    const int*  __restrict__ neg_idx,    // (T-1, N, BS)
    float* __restrict__ alpha_out,       // (BS)
    float* __restrict__ loss_out)        // scalar
{
    __shared__ float partial[8];
    __shared__ float logits_s[NNEG + 1];

    const int t = blockIdx.x;            // 0..T-2
    const int b = blockIdx.y;            // 0..BS-1
    const int d = threadIdx.x;           // 0..255

    const float wc = WCf[((size_t)b * TT + t) * DD + d];

    for (int c = 0; c <= NNEG; ++c) {
        const int brow = (c == 0) ? b : neg_idx[((size_t)t * NNEG + (c - 1)) * BS + b];
        float v = wc * P[((size_t)brow * TT + (t + 1)) * DD + d];
        #pragma unroll
        for (int off = 16; off > 0; off >>= 1) v += __shfl_down(v, off, 32);
        if ((threadIdx.x & 31) == 0) partial[threadIdx.x >> 5] = v;
        __syncthreads();
        if (threadIdx.x == 0) {
            float s = 0.f;
            #pragma unroll
            for (int i = 0; i < 8; ++i) s += partial[i];
            logits_s[c] = s * (1.f / (float)DD);
        }
        __syncthreads();
    }

    if (threadIdx.x == 0) {
        float mx = logits_s[0];
        for (int c = 1; c <= NNEG; ++c) mx = fmaxf(mx, logits_s[c]);
        float se = 0.f;
        for (int c = 0; c <= NNEG; ++c) se += __expf(logits_s[c] - mx);
        const float lse = mx + __logf(se);
        const float pos_ce = logits_s[0] - lse;
        atomicAdd(loss_out, -pos_ce * (1.f / (float)(TT - 1)));
        if (t == TT - 2) alpha_out[b] = logits_s[0];
    }
}

// ---------------------------------------------------------------------------
// y[b] = gelu(H[b,T-1] @ Wp1^T) @ Wp2^T ; one WG per b (tiny)
// ---------------------------------------------------------------------------
__global__ __launch_bounds__(256) void y_head_kernel(
    const float* __restrict__ Hf,        // (BS,T,HID) fp32 (d_out H region)
    const float* __restrict__ Wp1,       // (HID, HID)
    const float* __restrict__ Wp2,       // (1, HID)
    float* __restrict__ yout)            // (BS)
{
    __shared__ float partial[8];
    const int b = blockIdx.x;
    const int g = threadIdx.x;
    const float* hrow = Hf + ((size_t)b * TT + (TT - 1)) * HID;

    float acc = 0.f;
    for (int k = 0; k < HID; ++k) acc += hrow[k] * Wp1[(size_t)g * HID + k];
    float v = gelu_exact(acc) * Wp2[g];

    #pragma unroll
    for (int off = 16; off > 0; off >>= 1) v += __shfl_down(v, off, 32);
    if ((threadIdx.x & 31) == 0) partial[threadIdx.x >> 5] = v;
    __syncthreads();
    if (threadIdx.x == 0) {
        float s = 0.f;
        #pragma unroll
        for (int i = 0; i < 8; ++i) s += partial[i];
        yout[b] = s;
    }
}

// ---------------------------------------------------------------------------
// Host-side orchestration
// ---------------------------------------------------------------------------
extern "C" void kernel_launch(void* const* d_in, const int* in_sizes, int n_in,
                              void* d_out, int out_size, void* d_ws, size_t ws_size,
                              hipStream_t stream) {
    (void)in_sizes; (void)n_in; (void)out_size; (void)ws_size;

    const float* P    = (const float*)d_in[0];
    const float* Wih0 = (const float*)d_in[1];
    const float* Whh0 = (const float*)d_in[2];
    const float* bih0 = (const float*)d_in[3];
    const float* bhh0 = (const float*)d_in[4];
    const float* Wih1 = (const float*)d_in[5];
    const float* Whh1 = (const float*)d_in[6];
    const float* bih1 = (const float*)d_in[7];
    const float* bhh1 = (const float*)d_in[8];
    const float* W    = (const float*)d_in[9];
    const float* Wp1  = (const float*)d_in[10];
    const float* Wp2  = (const float*)d_in[11];
    const float* Wr1  = (const float*)d_in[12];
    const float* Wr2  = (const float*)d_in[13];
    const int*   neg  = (const int*)d_in[14];

    // d_out regions (return order: H, y, R, alpha, loss)
    float* out    = (float*)d_out;
    float* Hout   = out;                                   // BS*T*HID
    float* yout   = out + (size_t)BS * TT * HID;           // BS
    float* Rout   = yout + BS;                             // BS*T*D
    float* alpha  = Rout + (size_t)BS * TT * DD;           // BS
    float* loss   = alpha + BS;                            // 1

    // workspace layout
    char* ws = (char*)d_ws;
    size_t off = 0;
    auto take = [&](size_t bytes) { char* p = ws + off; off += (bytes + 255) & ~(size_t)255; return p; };
    __hip_bfloat16* Pbf    = (__hip_bfloat16*)take((size_t)BS * TT * DD * 2);
    __hip_bfloat16* H1bf   = (__hip_bfloat16*)take((size_t)BS * TT * HID * 2);
    __hip_bfloat16* Hbf    = (__hip_bfloat16*)take((size_t)BS * TT * HID * 2);
    float*          WCf    = (float*)take((size_t)BS * TT * DD * 4);
    __hip_bfloat16* Wih0bf = (__hip_bfloat16*)take((size_t)3 * HID * DD * 2);
    __hip_bfloat16* Whh0bf = (__hip_bfloat16*)take((size_t)3 * HID * HID * 2);
    __hip_bfloat16* Wih1bf = (__hip_bfloat16*)take((size_t)3 * HID * HID * 2);
    __hip_bfloat16* Whh1bf = (__hip_bfloat16*)take((size_t)3 * HID * HID * 2);
    __hip_bfloat16* Wbf    = (__hip_bfloat16*)take((size_t)DD * HID * 2);
    __hip_bfloat16* Wr1bf  = (__hip_bfloat16*)take((size_t)HID * HID * 2);
    __hip_bfloat16* Wr2bf  = (__hip_bfloat16*)take((size_t)DD * HID * 2);

    // 1) bf16 conversions
    cvt_f32_to_bf16<<<1024, 256, 0, stream>>>(P,    Pbf,    BS * TT * DD);
    cvt_f32_to_bf16<<<192,  256, 0, stream>>>(Wih0, Wih0bf, 3 * HID * DD);
    cvt_f32_to_bf16<<<192,  256, 0, stream>>>(Whh0, Whh0bf, 3 * HID * HID);
    cvt_f32_to_bf16<<<192,  256, 0, stream>>>(Wih1, Wih1bf, 3 * HID * HID);
    cvt_f32_to_bf16<<<192,  256, 0, stream>>>(Whh1, Whh1bf, 3 * HID * HID);
    cvt_f32_to_bf16<<<64,   256, 0, stream>>>(W,    Wbf,    DD * HID);
    cvt_f32_to_bf16<<<64,   256, 0, stream>>>(Wr1,  Wr1bf,  HID * HID);
    cvt_f32_to_bf16<<<64,   256, 0, stream>>>(Wr2,  Wr2bf,  DD * HID);

    // 2) GRU layer 0: P -> H1 (bf16 only)
    gru_layer_kernel<<<BS / 16, 256, 0, stream>>>(Pbf, Wih0bf, Whh0bf, bih0, bhh0, H1bf, nullptr);
    // 3) GRU layer 1: H1 -> H (bf16 + fp32 output)
    gru_layer_kernel<<<BS / 16, 256, 0, stream>>>(H1bf, Wih1bf, Whh1bf, bih1, bhh1, Hbf, Hout);

    // 4) WC = H @ W^T (all t; t=T-1 unused downstream)
    gemm_nt_kernel<<<(BS * TT) / 16, 256, 0, stream>>>(Hbf, Wbf, WCf);

    // 5) R = gelu(H @ Wr1^T) @ Wr2^T
    r_proj_kernel<<<(BS * TT) / 16, 256, 0, stream>>>(Hbf, Wr1bf, Wr2bf, Rout);

    // 6) contrastive loss + alpha
    zero_scalar<<<1, 1, 0, stream>>>(loss);
    contrast_kernel<<<dim3(TT - 1, BS), 256, 0, stream>>>(WCf, P, neg, alpha, loss);

    // 7) y head
    y_head_kernel<<<BS, 256, 0, stream>>>(Hout, Wp1, Wp2, yout);
}
